// DIN_PTCR_36739150250469
// MI455X (gfx1250) — matrix operand order repaired and checked by reading, not compiled, not run on hardware
//
#include <hip/hip_runtime.h>

typedef __attribute__((ext_vector_type(16))) _Float16 v16h;
typedef __attribute__((ext_vector_type(8)))  _Float16 v8h;
typedef __attribute__((ext_vector_type(8)))  float    v8f;

#define BB   4096
#define LL   200
#define LF   100
#define EE   64
#define PP   32
#define HH   32
#define IFD  128
#define UFD  128
#define TP   3168   // P + PNET

// ---------------------------------------------------------------------------
// Weight convert: W [K x N] f32 row-major -> WT [Npad x Kpad] f16 (transposed,
// zero padded).  WT[n*Kpad + k] = W[k*N + n].
// ---------------------------------------------------------------------------
__global__ __launch_bounds__(256) void convert_wT(
    const float* __restrict__ W, _Float16* __restrict__ WT,
    int K, int N, int Kpad, int Npad)
{
    int idx = blockIdx.x * 256 + threadIdx.x;
    int total = Kpad * Npad;
    if (idx >= total) return;
    int n = idx / Kpad, k = idx - n * Kpad;
    float v = (n < N && k < K) ? W[(size_t)k * N + n] : 0.0f;
    WT[idx] = (_Float16)v;
}

// ---------------------------------------------------------------------------
// Generic WMMA GEMM:  C[M x ldc] = act(A[M x K] @ W + bias)
//   A    : f32, row-major, stride K (K multiple of 32, templated)
//   WT   : f16, [ldc x K] (transposed weights, padded rows zeroed)
//   bias : f32[Nvalid] or nullptr
// Block = 128 threads (4 waves); each wave owns 16 rows, loops all N tiles.
// A fragments are converted to f16 once and held in registers for all tiles.
// ACT: 0 = none, 1 = relu, 2 = sigmoid.
// ---------------------------------------------------------------------------
template<int K, int ACT>
__global__ __launch_bounds__(128) void gemm_wmma(
    const float* __restrict__ A, const _Float16* __restrict__ WT,
    const float* __restrict__ bias, int Nvalid,
    float* __restrict__ C, int ldc)
{
    constexpr int KS = K / 32;
    const int wave = threadIdx.x >> 5;
    const int lane = threadIdx.x & 31;
    const int g    = lane >> 4;           // lane group 0/1
    const int hl   = lane & 15;
    const int m0   = blockIdx.x * 64 + wave * 16;

    // A fragments (16x32 f16 per k-step), same k order as B (line 80):
    // lane half g holds K{g*16 .. g*16+15} contiguously
    const float* arow = A + (size_t)(m0 + hl) * K;
    v16h af[KS];
#pragma unroll
    for (int ks = 0; ks < KS; ++ks) {
        const float* p = arow + ks * 32;
        v16h a;
#pragma unroll
        for (int i = 0; i < 8; ++i) {
            a[i]     = (_Float16)p[g * 16 + i];
            a[8 + i] = (_Float16)p[g * 16 + 8 + i];
        }
        af[ks] = a;
    }

    const int ntiles = ldc >> 4;
    for (int nt = 0; nt < ntiles; ++nt) {
        const int col = nt * 16 + hl;
        const _Float16* wcol = WT + (size_t)col * K;
        if (nt + 1 < ntiles)
            __builtin_prefetch(wcol + (size_t)16 * K, 0, 0);  // next N tile
        v8f c = {};
#pragma unroll
        for (int ks = 0; ks < KS; ++ks) {
            v16h bf = *(const v16h*)(wcol + ks * 32 + g * 16);
            c = __builtin_amdgcn_wmma_f32_16x16x32_f16(
                    false, af[ks], false, bf, (short)0, c, false, false);
        }
        float bv = 0.0f;
        if (bias != nullptr && col < Nvalid) bv = bias[col];
#pragma unroll
        for (int r = 0; r < 8; ++r) {
            float v = c[r] + bv;
            if (ACT == 1) v = fmaxf(v, 0.0f);
            if (ACT == 2) v = 1.0f / (1.0f + __expf(-v));
            if (col >= Nvalid) v = 0.0f;       // keep pad columns zero
            C[(size_t)(m0 + g * 8 + r) * ldc + col] = v;
        }
    }
}

// ---------------------------------------------------------------------------
// Attention + embedding assembly.  One block (256 threads) per sample.
// History embeddings are gathered straight into LDS with
// GLOBAL_LOAD_ASYNC_TO_LDS_B128 (ASYNCcnt-tracked, no VGPR round-trip),
// then masked softmax attention pooling, final embeds, and masked-mean
// prompt_input.
// ---------------------------------------------------------------------------
__global__ __launch_bounds__(256) void attention_kernel(
    const float* __restrict__ item_emb, const float* __restrict__ user_emb,
    const int* __restrict__ tgt_id, const int* __restrict__ hist_id,
    const int* __restrict__ hist_len, const int* __restrict__ uid,
    const int* __restrict__ pos_id, const unsigned char* __restrict__ pos_mask,
    const float* __restrict__ item_feat_e, const float* __restrict__ user_feat_e,
    float* __restrict__ final_user, float* __restrict__ final_item,
    float* __restrict__ prompt_input)
{
    __shared__ __align__(16) float s_hist[LL * EE];
    __shared__ float s_tgt[EE];
    __shared__ float s_attn[256];
    __shared__ float s_red[256];
    __shared__ float s_pool[4][EE];

    const int b   = blockIdx.x;
    const int tid = threadIdx.x;

    const int tgt = tgt_id[b];
    if (tid < EE) s_tgt[tid] = item_emb[(size_t)tgt * EE + tid];

    // async gather: 200 rows x 64 f32 = 3200 x 16B chunks directly into LDS
    for (int i = tid; i < LL * 16; i += 256) {
        const int l = i >> 4;                 // history slot
        const int c = (i & 15) << 2;          // float offset within row
        const float* src =
            item_emb + (size_t)hist_id[b * LL + l] * EE + c;
        unsigned dst = (unsigned)(uintptr_t)(s_hist + l * EE + c);
        asm volatile("global_load_async_to_lds_b128 %0, %1, off"
                     :: "v"(dst), "v"(src) : "memory");
    }
    asm volatile("s_wait_asynccnt 0" ::: "memory");
    __syncthreads();

    const int hlen = hist_len[b];
    float sc = -1e9f;
    if (tid < LL && tid < hlen) {
        float acc = 0.0f;
        for (int e = 0; e < EE; ++e) acc += s_tgt[e] * s_hist[tid * EE + e];
        sc = acc * 0.125f;                       // 1/sqrt(64)
    }
    s_red[tid] = sc;
    __syncthreads();
    for (int s = 128; s > 0; s >>= 1) {
        if (tid < s) s_red[tid] = fmaxf(s_red[tid], s_red[tid + s]);
        __syncthreads();
    }
    const float mx = s_red[0];
    __syncthreads();

    float ex = (tid < LL && tid < hlen) ? __expf(sc - mx) : 0.0f;
    s_attn[tid] = ex;
    s_red[tid]  = ex;
    __syncthreads();
    for (int s = 128; s > 0; s >>= 1) {
        if (tid < s) s_red[tid] += s_red[tid + s];
        __syncthreads();
    }
    const float inv = 1.0f / s_red[0];
    __syncthreads();

    // pooled[e] = inv * sum_l attn[l]*hist[l][e]   (4 chunks x 64 e)
    const int e  = tid & 63;
    const int ch = tid >> 6;
    float p = 0.0f;
    for (int l = ch; l < LL; l += 4) p += s_attn[l] * s_hist[l * EE + e];
    s_pool[ch][e] = p;
    __syncthreads();
    if (tid < EE) {
        float pooled = (s_pool[0][tid] + s_pool[1][tid] +
                        s_pool[2][tid] + s_pool[3][tid]) * inv;
        final_user[(size_t)b * EE + tid] =
            user_emb[(size_t)uid[b] * EE + tid] +
            user_feat_e[(size_t)b * EE + tid] + pooled;
        final_item[(size_t)b * EE + tid] =
            s_tgt[tid] + item_feat_e[(size_t)b * EE + tid];
    }

    // prompt_input: masked mean of pos feedback embeddings
    float m = (tid < LF) ? (pos_mask[b * LF + tid] ? 1.0f : 0.0f) : 0.0f;
    s_red[tid] = m;
    __syncthreads();
    for (int s = 128; s > 0; s >>= 1) {
        if (tid < s) s_red[tid] += s_red[tid + s];
        __syncthreads();
    }
    const float cnt = s_red[0];
    __syncthreads();

    float q = 0.0f;
    for (int l = ch; l < LF; l += 4) {
        if (pos_mask[b * LF + l])
            q += item_emb[(size_t)pos_id[b * LF + l] * EE + e];
    }
    s_pool[ch][e] = q;
    __syncthreads();
    if (tid < EE) {
        prompt_input[(size_t)b * EE + tid] =
            (s_pool[0][tid] + s_pool[1][tid] + s_pool[2][tid] + s_pool[3][tid]) / cnt;
    }
}

// ---------------------------------------------------------------------------
// Per-sample prompt MLP with WMMA.  One block (64 threads = 2 waves) per
// sample: wave 0 = pos feedback, wave 1 = neg feedback.
//   layer1: relu( fb[112x64] @ w1[64x32] + b1 )   (rows 100..111 zero pad)
//   layer2:        h[112x32] @ w2[32x32] + b2
//   pe[p] = sum over masked rows; loss = softplus(-(pos-neg))
// ---------------------------------------------------------------------------
__global__ __launch_bounds__(64) void prompt_mlp_kernel(
    const float* __restrict__ item_emb, const float* __restrict__ tp,
    const int* __restrict__ pos_id, const unsigned char* __restrict__ pos_mask,
    const int* __restrict__ neg_id, const unsigned char* __restrict__ neg_mask,
    float* __restrict__ pos_pe_out, float* __restrict__ loss_out)
{
    __shared__ __align__(32) _Float16 s_w1T[HH * EE];        // [h][e]
    __shared__ __align__(32) _Float16 s_w2T[PP * HH];        // [p][h]
    __shared__ __align__(32) _Float16 s_A[2][112 * EE];      // f16 fb tiles
    __shared__ __align__(32) _Float16 s_h[2][112 * HH];      // layer1 out
    __shared__ float s_b1[HH], s_b2[PP];
    __shared__ float s_pe[2 * PP];

    const int b    = blockIdx.x;
    const int tid  = threadIdx.x;
    const int wave = tid >> 5;
    const int lane = tid & 31;
    const int g    = lane >> 4;
    const int hl   = lane & 15;

    const float* tpb = tp + (size_t)b * TP;
    // w1[e][h] = tpb[32 + e*32 + h]  ->  s_w1T[h*64 + e]
    for (int i = tid; i < EE * HH; i += 64) {
        int e = i >> 5, h = i & 31;
        s_w1T[h * EE + e] = (_Float16)tpb[PP + i];
    }
    if (tid < HH) s_b1[tid] = tpb[PP + 2048 + tid];
    // w2[h][p] = tpb[32 + 2080 + h*32 + p] -> s_w2T[p*32 + h]
    for (int i = tid; i < HH * PP; i += 64) {
        int h = i >> 5, p = i & 31;
        s_w2T[p * HH + h] = (_Float16)tpb[PP + 2080 + i];
    }
    if (tid < PP) s_b2[tid] = tpb[PP + 3104 + tid];
    if (tid < 2 * PP) s_pe[tid] = 0.0f;

    const int* ids = wave ? neg_id : pos_id;
    const unsigned char* msk = wave ? neg_mask : pos_mask;

    // stage feedback embeddings (f32 -> f16), pad rows 100..111 with zeros
    _Float16* sA = s_A[wave];
    for (int i = lane; i < 112 * EE; i += 32) {
        int l = i >> 6, e = i & 63;
        float v = 0.0f;
        if (l < LF) v = item_emb[(size_t)ids[b * LF + l] * EE + e];
        sA[i] = (_Float16)v;
    }
    __syncthreads();

    // ---- layer 1: 7 M-tiles x 2 N-tiles x 2 K-steps -----------------------
    _Float16* sH = s_h[wave];
    for (int mt = 0; mt < 7; ++mt) {
        const _Float16* rowp = sA + (mt * 16 + hl) * EE;
        v16h a0, a1;
#pragma unroll
        for (int i = 0; i < 8; ++i) {
            a0[i]     = rowp[g * 16 + i];
            a0[8 + i] = rowp[g * 16 + 8 + i];
            a1[i]     = rowp[32 + g * 16 + i];
            a1[8 + i] = rowp[32 + g * 16 + 8 + i];
        }
#pragma unroll
        for (int nt = 0; nt < 2; ++nt) {
            const int hc = nt * 16 + hl;
            v16h b0 = *(const v16h*)(s_w1T + hc * EE + g * 16);
            v16h b1 = *(const v16h*)(s_w1T + hc * EE + 32 + g * 16);
            v8f c = {};
            c = __builtin_amdgcn_wmma_f32_16x16x32_f16(false, a0, false, b0, (short)0, c, false, false);
            c = __builtin_amdgcn_wmma_f32_16x16x32_f16(false, a1, false, b1, (short)0, c, false, false);
            const float bb = s_b1[hc];
#pragma unroll
            for (int r = 0; r < 8; ++r) {
                float v = fmaxf(c[r] + bb, 0.0f);
                sH[(mt * 16 + g * 8 + r) * HH + hc] = (_Float16)v;
            }
        }
    }
    __syncthreads();

    // ---- layer 2 + masked reduction ---------------------------------------
#pragma unroll
    for (int nt = 0; nt < 2; ++nt) {
        const int pc = nt * 16 + hl;
        v16h bf = *(const v16h*)(s_w2T + pc * HH + g * 16);
        const float b2v = s_b2[pc];
        float partial = 0.0f;
        for (int mt = 0; mt < 7; ++mt) {
            const _Float16* rowp = sH + (mt * 16 + hl) * HH;
            v16h a;
#pragma unroll
            for (int i = 0; i < 8; ++i) {
                a[i]     = rowp[g * 16 + i];
                a[8 + i] = rowp[g * 16 + 8 + i];
            }
            v8f c = {};
            c = __builtin_amdgcn_wmma_f32_16x16x32_f16(false, a, false, bf, (short)0, c, false, false);
#pragma unroll
            for (int r = 0; r < 8; ++r) {
                int m = mt * 16 + g * 8 + r;
                if (m < LF && msk[b * LF + m]) partial += c[r] + b2v;
            }
        }
        atomicAdd(&s_pe[wave * PP + pc], partial);
    }
    __syncthreads();

    if (tid < PP) {
        float pos = s_pe[tid], neg = s_pe[PP + tid];
        pos_pe_out[(size_t)b * PP + tid] = pos;
        float x  = -(pos - neg);
        float sp = fmaxf(x, 0.0f) + log1pf(__expf(-fabsf(x)));  // softplus
        loss_out[(size_t)b * PP + tid] = sp;
    }
}

// ---------------------------------------------------------------------------
// fusion_in[b] = concat(final_item[64], pos_pe[32], prompt_embed[32])
// ---------------------------------------------------------------------------
__global__ __launch_bounds__(256) void fusion_in_kernel(
    const float* __restrict__ final_item, const float* __restrict__ pos_pe,
    const float* __restrict__ tp, float* __restrict__ fin)
{
    int idx = blockIdx.x * 256 + threadIdx.x;
    if (idx >= BB * 128) return;
    int b = idx >> 7, c = idx & 127;
    float v;
    if (c < 64)       v = final_item[(size_t)b * EE + c];
    else if (c < 96)  v = pos_pe[(size_t)b * PP + (c - 64)];
    else              v = tp[(size_t)b * TP + (c - 96)];
    fin[idx] = v;
}

// ---------------------------------------------------------------------------
// output[b] = dot(final_user[b], fused_item[b])
// ---------------------------------------------------------------------------
__global__ __launch_bounds__(256) void final_dot_kernel(
    const float* __restrict__ fue, const float* __restrict__ fused,
    float* __restrict__ out)
{
    int b = blockIdx.x * 256 + threadIdx.x;
    if (b >= BB) return;
    float acc = 0.0f;
    for (int e = 0; e < EE; ++e)
        acc += fue[(size_t)b * EE + e] * fused[(size_t)b * EE + e];
    out[b] = acc;
}

// ---------------------------------------------------------------------------
extern "C" void kernel_launch(void* const* d_in, const int* in_sizes, int n_in,
                              void* d_out, int out_size, void* d_ws, size_t ws_size,
                              hipStream_t stream)
{
    (void)in_sizes; (void)n_in; (void)out_size; (void)ws_size;

    const float* item_emb  = (const float*)d_in[0];
    const float* user_emb  = (const float*)d_in[1];
    const float* W_if      = (const float*)d_in[2];
    const float* b_if      = (const float*)d_in[3];
    const float* W_uf      = (const float*)d_in[4];
    const float* b_uf      = (const float*)d_in[5];
    const float* Wp        = (const float*)d_in[6];
    const float* bp        = (const float*)d_in[7];
    const float* Wf1       = (const float*)d_in[8];
    const float* bf1       = (const float*)d_in[9];
    const float* Wf2       = (const float*)d_in[10];
    const float* bf2       = (const float*)d_in[11];
    const float* Wf3       = (const float*)d_in[12];
    const float* item_feat = (const float*)d_in[13];
    const float* user_feat = (const float*)d_in[14];
    const int*   uid       = (const int*)d_in[15];
    const int*   tgt_id    = (const int*)d_in[16];
    const int*   hist_id   = (const int*)d_in[17];
    const int*   hist_len  = (const int*)d_in[18];
    const int*   pos_id    = (const int*)d_in[19];
    const unsigned char* pos_mask = (const unsigned char*)d_in[20];
    const int*   neg_id    = (const int*)d_in[21];
    const unsigned char* neg_mask = (const unsigned char*)d_in[22];

    float* out  = (float*)d_out;            // [B] output
    float* loss = out + BB;                 // [B x 32] loss_pfpe

    // ---- workspace layout -------------------------------------------------
    char*  base = (char*)d_ws;
    size_t off  = 0;
    auto alloc = [&](size_t bytes) -> char* {
        char* p = base + off;
        off += (bytes + 255) & ~(size_t)255;
        return p;
    };
    _Float16* WT_if = (_Float16*)alloc((size_t)64 * 128 * 2);
    _Float16* WT_uf = (_Float16*)alloc((size_t)64 * 128 * 2);
    _Float16* WT_p  = (_Float16*)alloc((size_t)TP * 64 * 2);
    _Float16* WT_f1 = (_Float16*)alloc((size_t)224 * 128 * 2);
    _Float16* WT_f2 = (_Float16*)alloc((size_t)96 * 224 * 2);
    _Float16* WT_f3 = (_Float16*)alloc((size_t)64 * 96 * 2);
    float* item_fe  = (float*)alloc((size_t)BB * EE * 4);
    float* user_fe  = (float*)alloc((size_t)BB * EE * 4);
    float* prompt_in= (float*)alloc((size_t)BB * EE * 4);
    float* fin_user = (float*)alloc((size_t)BB * EE * 4);
    float* fin_item = (float*)alloc((size_t)BB * EE * 4);
    float* tp_buf   = (float*)alloc((size_t)BB * TP * 4);
    float* pos_pe   = (float*)alloc((size_t)BB * PP * 4);
    float* fusion_in= (float*)alloc((size_t)BB * 128 * 4);
    float* h1       = (float*)alloc((size_t)BB * 224 * 4);
    float* h2       = (float*)alloc((size_t)BB * 96 * 4);
    float* fused    = (float*)alloc((size_t)BB * 64 * 4);

    auto cgrid = [](int total) { return (total + 255) / 256; };

    // 1) convert weights to transposed padded f16
    convert_wT<<<cgrid(64 * 128), 256, 0, stream>>>(W_if, WT_if, 128, 64, 128, 64);
    convert_wT<<<cgrid(64 * 128), 256, 0, stream>>>(W_uf, WT_uf, 128, 64, 128, 64);
    convert_wT<<<cgrid(TP * 64), 256, 0, stream>>>(Wp,   WT_p,  64, TP, 64, TP);
    convert_wT<<<cgrid(224 * 128), 256, 0, stream>>>(Wf1, WT_f1, 128, 200, 128, 224);
    convert_wT<<<cgrid(96 * 224), 256, 0, stream>>>(Wf2,  WT_f2, 200, 80, 224, 96);
    convert_wT<<<cgrid(64 * 96), 256, 0, stream>>>(Wf3,   WT_f3, 80, 64, 96, 64);

    // 2) feature embedding nets (sigmoid)
    gemm_wmma<128, 2><<<BB / 64, 128, 0, stream>>>(item_feat, WT_if, b_if, 64, item_fe, 64);
    gemm_wmma<128, 2><<<BB / 64, 128, 0, stream>>>(user_feat, WT_uf, b_uf, 64, user_fe, 64);

    // 3) attention + embedding assembly + prompt input (async LDS gather)
    attention_kernel<<<BB, 256, 0, stream>>>(
        item_emb, user_emb, tgt_id, hist_id, hist_len, uid,
        pos_id, pos_mask, item_fe, user_fe, fin_user, fin_item, prompt_in);

    // 4) total_prompt = relu(prompt_in @ Wp + bp)
    gemm_wmma<64, 1><<<BB / 64, 128, 0, stream>>>(prompt_in, WT_p, bp, TP, tp_buf, TP);

    // 5) per-sample prompt MLP (pos/neg), pe's + loss
    prompt_mlp_kernel<<<BB, 64, 0, stream>>>(
        item_emb, tp_buf, pos_id, pos_mask, neg_id, neg_mask, pos_pe, loss);

    // 6) fusion MLP
    fusion_in_kernel<<<cgrid(BB * 128), 256, 0, stream>>>(fin_item, pos_pe, tp_buf, fusion_in);
    gemm_wmma<128, 1><<<BB / 64, 128, 0, stream>>>(fusion_in, WT_f1, bf1, 200, h1, 224);
    gemm_wmma<224, 1><<<BB / 64, 128, 0, stream>>>(h1, WT_f2, bf2, 80, h2, 96);
    gemm_wmma<96, 0><<<BB / 64, 128, 0, stream>>>(h2, WT_f3, nullptr, 64, fused, 64);

    // 7) final dot product
    final_dot_kernel<<<cgrid(BB), 256, 0, stream>>>(fin_user, fused, out);
}
